// AttentionBlock_33466385170746
// MI455X (gfx1250) — matrix operand order repaired and checked by reading, not compiled, not run on hardware
//
#include <hip/hip_runtime.h>

// ---------------------------------------------------------------------------
// GPT-2 attention block for MI455X (gfx1250, wave32, WMMA).
// B=2, M=2048, N=1024, H=16, d=64.  All GEMMs use v_wmma_f32_16x16x32_f16
// (f16 inputs, f32 accumulation). Attention is flash-style so the [M,M]
// score matrix is never materialized in HBM.
//
// GEMMs: 64(M) x 128(N) workgroup tile, 8 waves as 2x4, each wave 32x32
// (4 accumulators), K staged 64-deep in LDS -> 8 WMMAs per barrier round.
// global_prefetch_b8 issued for the next K tile while computing this one.
// ---------------------------------------------------------------------------

typedef __attribute__((ext_vector_type(16))) _Float16 v16h;
typedef __attribute__((ext_vector_type(8)))  float    v8f;

#define BATCH   2
#define SEQ     2048
#define NMODEL  1024
#define NHEAD   16
#define HDIM    64
#define ROWS    (BATCH * SEQ)      // 4096
#define N_QKV   (3 * NMODEL)       // 3072
#define MASKNEG (-1.0e30f)
#define LDT     66                 // padded f16 row stride (33 dwords)

// ---- WMMA helper ----------------------------------------------------------
__device__ __forceinline__ v8f wmma16(v16h a, v16h b, v8f c) {
  // (neg_a, A, neg_b, B, c_mod, C, reuse_a, reuse_b)
  return __builtin_amdgcn_wmma_f32_16x16x32_f16(false, a, false, b,
                                                (short)0, c, false, false);
}

// A-fragment (16x32 f16, M x K): lane = row (lane&15), g = lane>>4.
// VGPR j in 0..3 holds K = 8g+2j, 8g+2j+1 ; j in 4..7 holds K = 16+8g+2(j-4).
// p points at the 32-element K-run for this lane's row.
__device__ __forceinline__ v16h ld_fragA(const _Float16* p, int g) {
  v16h f;
#pragma unroll
  for (int j = 0; j < 8; ++j) {
    int kb = ((j < 4) ? (2 * j) : (16 + 2 * (j - 4))) + g * 8;
    f[2 * j]     = p[kb];
    f[2 * j + 1] = p[kb + 1];
  }
  return f;
}

// B-fragment (32x16 f16, K x N): lane = column (lane&15), g = lane>>4,
// element i holds K = 8*g + i (i<8), 16 + 8*g + (i-8) (i>=8).  p points at Bt[col][k0..k0+31]
// where Bt is the transposed (column-major) tile: Bt[n][k] = B[k][n].
__device__ __forceinline__ v16h ld_fragB(const _Float16* p, int g) {
  v16h f;
#pragma unroll
  for (int i = 0; i < 16; ++i) f[i] = p[((i < 8) ? i : (i + 8)) + g * 8];
  return f;
}

// ---------------------------------------------------------------------------
// Kernel 1: fused QKV projection.  x[4096,1024] * w_attn[1024,3072] + b_attn.
// Epilogue converts to f16 and scatters into attention-friendly layouts:
//   Qh[b*H+h][m][d]  (pre-scaled by 1/sqrt(d))
//   Kh[b*H+h][m][d]
//   Vt[b*H+h][d][m]  (transposed -> B-fragment-ready for P@V)
// ---------------------------------------------------------------------------
__global__ __launch_bounds__(256) void qkv_gemm_k(
    const float* __restrict__ x, const float* __restrict__ w,
    const float* __restrict__ bias,
    _Float16* __restrict__ Qh, _Float16* __restrict__ Kh,
    _Float16* __restrict__ Vt) {
  __shared__ _Float16 As[64 * LDT];    // [row][k]   64 x 64
  __shared__ _Float16 Bt[128 * LDT];   // [col][k]  128 x 64 (transposed)

  const int t    = threadIdx.x;
  const int lane = t & 31;
  const int wid  = t >> 5;
  const int wr   = wid & 1;            // 2 row groups x 32 rows
  const int wc   = wid >> 1;           // 4 col groups x 32 cols
  const int g    = lane >> 4;
  const int l16  = lane & 15;
  const int rowBase = blockIdx.y * 64;
  const int colBase = blockIdx.x * 128;

  v8f c00 = {}, c01 = {}, c10 = {}, c11 = {};

  for (int k0 = 0; k0 < NMODEL; k0 += 64) {
    // prefetch next K tile into L2/WGP$ while this one is computed
    if (k0 + 64 < NMODEL) {
      __builtin_prefetch(&x[(size_t)(rowBase + (t >> 2)) * NMODEL + k0 + 64 +
                            ((t & 3) << 4)], 0, 1);
      __builtin_prefetch(&w[(size_t)(k0 + 64 + (t >> 2)) * N_QKV + colBase +
                            ((t & 3) << 5)], 0, 1);
    }
    // stage A (64x64, f32 -> f16): 16 elements / thread, 256B rows coalesced
#pragma unroll
    for (int i = 0; i < 16; ++i) {
      int idx = t + i * 256;
      int r = idx >> 6, cc = idx & 63;
      As[r * LDT + cc] = (_Float16)x[(size_t)(rowBase + r) * NMODEL + k0 + cc];
    }
    // stage B transposed (64k x 128n): Bt[n][k] = w[(k0+k)*3072 + colBase+n]
#pragma unroll 8
    for (int i = 0; i < 32; ++i) {
      int idx = t + i * 256;
      int k = idx >> 7, n = idx & 127;
      Bt[n * LDT + k] = (_Float16)w[(size_t)(k0 + k) * N_QKV + colBase + n];
    }
    __syncthreads();

#pragma unroll
    for (int cc = 0; cc < 64; cc += 32) {
      v16h a0 = ld_fragA(&As[(wr * 32 + l16) * LDT + cc], g);
      v16h a1 = ld_fragA(&As[(wr * 32 + 16 + l16) * LDT + cc], g);
      v16h b0 = ld_fragB(&Bt[(wc * 32 + l16) * LDT + cc], g);
      v16h b1 = ld_fragB(&Bt[(wc * 32 + 16 + l16) * LDT + cc], g);
      c00 = wmma16(a0, b0, c00);
      c01 = wmma16(a0, b1, c01);
      c10 = wmma16(a1, b0, c10);
      c11 = wmma16(a1, b1, c11);
    }
    __syncthreads();
  }

  // epilogue: bias, f16 convert, scatter into Q/K/V layouts
#pragma unroll
  for (int sr = 0; sr < 2; ++sr) {
#pragma unroll
    for (int sc = 0; sc < 2; ++sc) {
      v8f c = sr ? (sc ? c11 : c10) : (sc ? c01 : c00);
      int col  = colBase + wc * 32 + sc * 16 + l16;
      int part = col >> 10;   // 0:q 1:k 2:v (uniform across the 16-col subtile)
      int n    = col & 1023;
      int h    = n >> 6, dd = n & 63;
      float bv = bias[col];
#pragma unroll
      for (int e = 0; e < 8; ++e) {
        int r = rowBase + wr * 32 + sr * 16 + e + 8 * g;
        int b = r >> 11, m = r & 2047;
        int bh = b * NHEAD + h;
        float v = c[e] + bv;
        if (part == 0) {
          Qh[((size_t)bh * SEQ + m) * HDIM + dd] = (_Float16)(v * 0.125f);
        } else if (part == 1) {
          Kh[((size_t)bh * SEQ + m) * HDIM + dd] = (_Float16)v;
        } else {
          Vt[((size_t)bh * HDIM + dd) * SEQ + m] = (_Float16)v;
        }
      }
    }
  }
}

// ---------------------------------------------------------------------------
// Kernel 2: flash attention.  One WG (4 waves) per 64 query rows per (b,h).
// Streams 64-key blocks through LDS; online softmax; P@V via WMMA with P
// round-tripped through a per-wave LDS tile (C-layout -> A-fragment layout).
// ---------------------------------------------------------------------------
__global__ __launch_bounds__(128) void flash_attn_k(
    const _Float16* __restrict__ Qh, const _Float16* __restrict__ Kh,
    const _Float16* __restrict__ Vt, _Float16* __restrict__ ctx) {
  __shared__ _Float16 Ks[64 * LDT];        // K block: [key][d]
  __shared__ _Float16 Vs[64 * LDT];        // V block: [d][key]
  __shared__ _Float16 Ps[4 * 16 * LDT];    // per-wave probability tiles

  const int t    = threadIdx.x;
  const int lane = t & 31;
  const int wv   = t >> 5;                 // wave id 0..3
  const int g    = lane >> 4;
  const int l16  = lane & 15;
  const int bh   = blockIdx.y;             // 0..31
  const int qb   = blockIdx.x * 64;
  const int b    = bh >> 4, h = bh & 15;
  const int qrow0 = qb + wv * 16;

  const _Float16* Qbase = Qh + (size_t)bh * SEQ * HDIM;
  const _Float16* Kbase = Kh + (size_t)bh * SEQ * HDIM;
  const _Float16* Vbase = Vt + (size_t)bh * HDIM * SEQ;

  // Q fragments for this wave's 16 rows, resident in registers for the loop
  v16h q0 = ld_fragA(Qbase + (size_t)(qrow0 + l16) * HDIM + 0,  g);
  v16h q1 = ld_fragA(Qbase + (size_t)(qrow0 + l16) * HDIM + 32, g);

  v8f o[4] = {{}, {}, {}, {}};
  float mrow[8], lrow[8];
#pragma unroll
  for (int e = 0; e < 8; ++e) { mrow[e] = MASKNEG; lrow[e] = 0.0f; }

  const int nkb = (qb >> 6) + 1;           // causal: blocks 0..qb/64
  for (int ib = 0; ib < nkb; ++ib) {
    const int kb = ib * 64;
    // prefetch the next key block while this one is staged/consumed
    if (ib + 1 < nkb) {
      int kn = kb + 64;
      __builtin_prefetch(Kbase + (size_t)(kn + (t >> 1)) * HDIM + ((t & 1) << 5), 0, 1);
      __builtin_prefetch(Vbase + (size_t)(t >> 1) * SEQ + kn + ((t & 1) << 5), 0, 1);
    }
    // ---- cooperative staging of K (64x64) and V (64x64) as dwords --------
    uint32_t* Kd = (uint32_t*)Ks;
    uint32_t* Vd = (uint32_t*)Vs;
#pragma unroll
    for (int i = 0; i < 16; ++i) {
      int idx = t + i * 128;               // 0..2047 dwords
      int r = idx >> 5, c = idx & 31;
      Kd[r * (LDT / 2) + c] = ((const uint32_t*)(Kbase + (size_t)(kb + r) * HDIM))[c];
      Vd[r * (LDT / 2) + c] = ((const uint32_t*)(Vbase + (size_t)r * SEQ + kb))[c];
    }
    __syncthreads();

    // ---- S = Q @ K^T : 16 q-rows x 64 keys (8 WMMAs) ----------------------
    v8f sacc[4] = {{}, {}, {}, {}};
#pragma unroll
    for (int tt = 0; tt < 2; ++tt) {
      v16h aq = tt ? q1 : q0;
#pragma unroll
      for (int ss = 0; ss < 4; ++ss) {
        v16h bk = ld_fragB(&Ks[(ss * 16 + l16) * LDT + tt * 32], g);
        sacc[ss] = wmma16(aq, bk, sacc[ss]);
      }
    }

    // ---- causal mask + online softmax (row = e + 8g, col = lane&15) -------
#pragma unroll
    for (int e = 0; e < 8; ++e) {
      const int q = qrow0 + e + 8 * g;
      float mblk = MASKNEG;
#pragma unroll
      for (int ss = 0; ss < 4; ++ss) {
        int kidx = kb + ss * 16 + l16;
        float v = sacc[ss][e];
        if (kidx > q) v = MASKNEG;
        sacc[ss][e] = v;
        mblk = fmaxf(mblk, v);
      }
#pragma unroll
      for (int off = 1; off < 16; off <<= 1)    // reduce within 16-lane group
        mblk = fmaxf(mblk, __shfl_xor(mblk, off, 32));
      float mnew  = fmaxf(mrow[e], mblk);
      float alpha = __expf(mrow[e] - mnew);
      float rsum  = 0.0f;
#pragma unroll
      for (int ss = 0; ss < 4; ++ss) {
        float p = __expf(sacc[ss][e] - mnew);
        sacc[ss][e] = p;
        rsum += p;
      }
#pragma unroll
      for (int off = 1; off < 16; off <<= 1)
        rsum += __shfl_xor(rsum, off, 32);
      lrow[e] = lrow[e] * alpha + rsum;
      mrow[e] = mnew;
      o[0][e] *= alpha; o[1][e] *= alpha; o[2][e] *= alpha; o[3][e] *= alpha;
    }

    // ---- C-layout -> A-layout via per-wave LDS tile -----------------------
    _Float16* Pw = &Ps[wv * 16 * LDT];
#pragma unroll
    for (int ss = 0; ss < 4; ++ss)
#pragma unroll
      for (int e = 0; e < 8; ++e)
        Pw[(e + 8 * g) * LDT + ss * 16 + l16] = (_Float16)sacc[ss][e];
    // same-wave LDS store->load is in-order; no workgroup barrier needed here

    // ---- O += P @ V : 16 q-rows x 64 d (8 WMMAs) --------------------------
#pragma unroll
    for (int tt = 0; tt < 2; ++tt) {
      v16h ap = ld_fragA(&Pw[l16 * LDT + tt * 32], g);
#pragma unroll
      for (int ss = 0; ss < 4; ++ss) {
        v16h bv = ld_fragB(&Vs[(ss * 16 + l16) * LDT + tt * 32], g);
        o[ss] = wmma16(ap, bv, o[ss]);
      }
    }
    __syncthreads();   // protect Ks/Vs before next block's staging
  }

  // ---- normalize and emit ctx as f16 [b][m][h*64+d] ------------------------
#pragma unroll
  for (int e = 0; e < 8; ++e) {
    float inv = 1.0f / lrow[e];
    int q = qrow0 + e + 8 * g;
#pragma unroll
    for (int ss = 0; ss < 4; ++ss)
      ctx[((size_t)(b * SEQ + q)) * NMODEL + h * HDIM + ss * 16 + l16] =
          (_Float16)(o[ss][e] * inv);
  }
}

// ---------------------------------------------------------------------------
// Kernel 3: output projection.  ctx(f16)[4096,1024] * w_proj[1024,1024] + b.
// ---------------------------------------------------------------------------
__global__ __launch_bounds__(256) void proj_gemm_k(
    const _Float16* __restrict__ ctx, const float* __restrict__ w,
    const float* __restrict__ bias, float* __restrict__ out) {
  __shared__ _Float16 As[64 * LDT];
  __shared__ _Float16 Bt[128 * LDT];

  const int t    = threadIdx.x;
  const int lane = t & 31;
  const int wid  = t >> 5;
  const int wr   = wid & 1;
  const int wc   = wid >> 1;
  const int g    = lane >> 4;
  const int l16  = lane & 15;
  const int rowBase = blockIdx.y * 64;
  const int colBase = blockIdx.x * 128;

  v8f c00 = {}, c01 = {}, c10 = {}, c11 = {};

  for (int k0 = 0; k0 < NMODEL; k0 += 64) {
    if (k0 + 64 < NMODEL) {
      __builtin_prefetch(ctx + (size_t)(rowBase + (t >> 2)) * NMODEL + k0 + 64 +
                         ((t & 3) << 4), 0, 1);
      __builtin_prefetch(&w[(size_t)(k0 + 64 + (t >> 2)) * NMODEL + colBase +
                            ((t & 3) << 5)], 0, 1);
    }
    // stage A (f16 already): dword copies, 8 / thread
    uint32_t* Ad = (uint32_t*)As;
#pragma unroll
    for (int i = 0; i < 8; ++i) {
      int idx = t + i * 256;
      int r = idx >> 5, c = idx & 31;
      Ad[r * (LDT / 2) + c] =
          ((const uint32_t*)(ctx + (size_t)(rowBase + r) * NMODEL + k0))[c];
    }
    // stage B transposed (64k x 128n), f32 -> f16
#pragma unroll 8
    for (int i = 0; i < 32; ++i) {
      int idx = t + i * 256;
      int k = idx >> 7, n = idx & 127;
      Bt[n * LDT + k] = (_Float16)w[(size_t)(k0 + k) * NMODEL + colBase + n];
    }
    __syncthreads();

#pragma unroll
    for (int cc = 0; cc < 64; cc += 32) {
      v16h a0 = ld_fragA(&As[(wr * 32 + l16) * LDT + cc], g);
      v16h a1 = ld_fragA(&As[(wr * 32 + 16 + l16) * LDT + cc], g);
      v16h b0 = ld_fragB(&Bt[(wc * 32 + l16) * LDT + cc], g);
      v16h b1 = ld_fragB(&Bt[(wc * 32 + 16 + l16) * LDT + cc], g);
      c00 = wmma16(a0, b0, c00);
      c01 = wmma16(a0, b1, c01);
      c10 = wmma16(a1, b0, c10);
      c11 = wmma16(a1, b1, c11);
    }
    __syncthreads();
  }

#pragma unroll
  for (int sr = 0; sr < 2; ++sr) {
#pragma unroll
    for (int sc = 0; sc < 2; ++sc) {
      v8f c = sr ? (sc ? c11 : c10) : (sc ? c01 : c00);
      int col = colBase + wc * 32 + sc * 16 + l16;
      float bv = bias[col];
#pragma unroll
      for (int e = 0; e < 8; ++e) {
        int r = rowBase + wr * 32 + sr * 16 + e + 8 * g;
        out[(size_t)r * NMODEL + col] = c[e] + bv;
      }
    }
  }
}

// ---------------------------------------------------------------------------
extern "C" void kernel_launch(void* const* d_in, const int* in_sizes, int n_in,
                              void* d_out, int out_size, void* d_ws, size_t ws_size,
                              hipStream_t stream) {
  const float* x      = (const float*)d_in[0];
  const float* w_attn = (const float*)d_in[1];
  const float* w_proj = (const float*)d_in[2];
  const float* b_attn = (const float*)d_in[3];
  const float* b_proj = (const float*)d_in[4];
  float*       out    = (float*)d_out;

  // workspace: Qh | Kh | Vt | ctx, each B*H*M*64 f16 = 8 MB (32 MB total)
  char* ws = (char*)d_ws;
  const size_t SEG = (size_t)BATCH * NHEAD * SEQ * HDIM * sizeof(_Float16);
  _Float16* Qh  = (_Float16*)(ws);
  _Float16* Kh  = (_Float16*)(ws + SEG);
  _Float16* Vt  = (_Float16*)(ws + 2 * SEG);
  _Float16* ctx = (_Float16*)(ws + 3 * SEG);

  qkv_gemm_k<<<dim3(N_QKV / 128, ROWS / 64), 256, 0, stream>>>(
      x, w_attn, b_attn, Qh, Kh, Vt);
  flash_attn_k<<<dim3(SEQ / 64, BATCH * NHEAD), 128, 0, stream>>>(
      Qh, Kh, Vt, ctx);
  proj_gemm_k<<<dim3(NMODEL / 128, ROWS / 64), 256, 0, stream>>>(
      ctx, w_proj, b_proj, out);
}
